// Frame_Level_Encoder_24515673325621
// MI455X (gfx1250) — matrix operand order repaired and checked by reading, not compiled, not run on hardware
//
#include <hip/hip_runtime.h>

// ---------------------------------------------------------------------------
// Fused frame-level encoder for MI455X (gfx1250, wave32, WMMA, bf16).
//   h1 = relu(x@W1+b1); h2 = relu(h1@W2+b2)
//   Q,K,V = h2@W{q,k,v}+b;  out = sigmoid(<Q,K>)*V;  LayerNorm;  y = out@Wo+bo
// Each wave owns 16 token rows end-to-end. Activations live in per-wave LDS
// buffers (bf16, A-fragment-friendly row-major); B-fragments stream from the
// repacked weight workspace with a depth-8 register prefetch pipeline so
// v_wmma never sits behind a cold s_wait_loadcnt 0.
// ---------------------------------------------------------------------------

typedef __attribute__((ext_vector_type(16))) __bf16        v16bf;
typedef __attribute__((ext_vector_type(8)))  float         v8f;
typedef __attribute__((ext_vector_type(4)))  unsigned int  u32x4;
typedef __attribute__((ext_vector_type(4)))  float         f32x4;

union Frag { u32x4 q[2]; unsigned u[8]; v16bf v; };

// ---- workspace layout (bf16 elements) -------------------------------------
constexpr int OFF_W1 = 0;                       // 512x512
constexpr int OFF_W2 = OFF_W1 + 512 * 512;      // 512x512
constexpr int OFF_WQ = OFF_W2 + 512 * 512;      // 512x256
constexpr int OFF_WK = OFF_WQ + 512 * 256;
constexpr int OFF_WV = OFF_WK + 512 * 256;
constexpr int OFF_WO = OFF_WV + 512 * 256;      // 256x512

// ---- helpers ---------------------------------------------------------------
__device__ inline v8f vzero() {
  v8f z = {0.f, 0.f, 0.f, 0.f, 0.f, 0.f, 0.f, 0.f};
  return z;
}

__device__ inline unsigned pk2(float a, float b) {  // 2x f32 -> packed bf16 (RNE)
  unsigned ua = __float_as_uint(a), ub = __float_as_uint(b);
  ua += 0x7FFFu + ((ua >> 16) & 1u);
  ub += 0x7FFFu + ((ub >> 16) & 1u);
  return (ua >> 16) | (ub & 0xFFFF0000u);
}

__device__ inline unsigned short f2bf(float a) {
  unsigned ua = __float_as_uint(a);
  ua += 0x7FFFu + ((ua >> 16) & 1u);
  return (unsigned short)(ua >> 16);
}

__device__ inline v8f wmma_bf16(v16bf a, v16bf b, v8f c) {
  return __builtin_amdgcn_wmma_f32_16x16x32_bf16(false, a, false, b,
                                                 (short)0, c, false, false);
}

// B-fragment: each (ktile,ntile) 32x16 tile is stored as 32 lanes x 16 bf16
// contiguous -> one 32B (2x global_load_b128) fetch per lane.
__device__ inline v16bf ldB(const unsigned short* __restrict__ Wf, int tile, int lane) {
  Frag f;
  const u32x4* p = (const u32x4*)(Wf + (size_t)tile * 512 + lane * 16);
  f.q[0] = p[0];
  f.q[1] = p[1];
  return f.v;
}

// A-fragment from a row-major [16 x stride] bf16 LDS activation buffer.
// Lanes 0-15 (row L) take K {0..7,16..23}; lanes 16-31 take K {8..15,24..31}.
__device__ inline v16bf ldA_act(const unsigned short* buf, int k, int lane, int stride) {
  const int rL = lane & 15, g = (lane >> 4) & 1;
  const unsigned short* p = buf + rL * stride + k * 32 + g * 8;
  Frag f;
  f.q[0] = *(const u32x4*)(p);
  f.q[1] = *(const u32x4*)(p + 16);
  return f.v;
}

// Store one 16x16 f32 C-layout accumulator tile (coltile ct) into a row-major
// bf16 activation buffer.
__device__ inline void stTile_act(unsigned short* buf, const v8f acc, int ct,
                                  int lane, int stride) {
  const int hi8 = (lane >= 16) ? 8 : 0, n = lane & 15;
  #pragma unroll
  for (int v = 0; v < 8; ++v)
    buf[(v + hi8) * stride + ct * 16 + n] = f2bf(acc[v]);
}

// Dense layer with ReLU: out[16 x NT*16] = relu(in[16 x 512] * W + bias).
// Depth-8 software pipeline: 8 A/B fragment slots, each refilled with the
// (k+8) fragment right after its WMMA consumes it.
template <int NT>
__device__ inline void dense_relu(const unsigned short* __restrict__ Wf,
                                  const float* __restrict__ bias,
                                  const unsigned short* inb, unsigned short* outb,
                                  int lane) {
  const int n0 = lane & 15;
  for (int ct = 0; ct < NT; ++ct) {
    // prefetch next coltile's weight tile into WGP$/L2 (global_prefetch_b8)
    if (ct + 1 < NT) {
      __builtin_prefetch(Wf + (size_t)(0 * NT + ct + 1) * 512 + lane * 16, 0, 0);
      __builtin_prefetch(Wf + (size_t)(8 * NT + ct + 1) * 512 + lane * 16, 0, 0);
    }
    v16bf a[8], b[8];
    #pragma unroll
    for (int k = 0; k < 8; ++k) b[k] = ldB(Wf, k * NT + ct, lane);
    #pragma unroll
    for (int k = 0; k < 8; ++k) a[k] = ldA_act(inb, k, lane, 512);
    v8f acc = vzero();
    #pragma unroll
    for (int k = 0; k < 8; ++k) {
      acc = wmma_bf16(a[k], b[k], acc);
      b[k] = ldB(Wf, (k + 8) * NT + ct, lane);   // refill slot with k+8
      a[k] = ldA_act(inb, k + 8, lane, 512);
    }
    #pragma unroll
    for (int k = 0; k < 8; ++k) acc = wmma_bf16(a[k], b[k], acc);
    const float bs = bias[ct * 16 + n0];
    #pragma unroll
    for (int v = 0; v < 8; ++v) {
      float t = acc[v] + bs;
      acc[v] = t > 0.f ? t : 0.f;
    }
    stTile_act(outb, acc, ct, lane, 512);
  }
}

// ---- weight repack: fp32 row-major [K,N] -> bf16 B-fragment order ----------
__global__ void prep_w(const float* __restrict__ W, unsigned short* __restrict__ Wf,
                       int N, int total) {
  int tid = blockIdx.x * blockDim.x + threadIdx.x;
  if (tid >= total) return;
  const int j    = tid & 15;
  const int lane = (tid >> 4) & 31;
  const int rest = tid >> 9;            // kt*(N/16) + nt
  const int NT   = N >> 4;
  const int nt   = rest % NT;
  const int kt   = rest / NT;
  const int col  = nt * 16 + (lane & 15);
  const int k    = kt * 32 + ((lane >> 4) << 3) + (j & 7) + ((j >> 3) << 4);  // lanes 16-31 -> K {8..15,24..31}
  unsigned bits = __float_as_uint(W[(size_t)k * N + col]);
  bits += 0x7FFFu + ((bits >> 16) & 1u);
  Wf[tid] = (unsigned short)(bits >> 16);
}

// ---- fused encoder ---------------------------------------------------------
__global__ __launch_bounds__(64, 1) void fused_encoder(
    const float* __restrict__ x,
    const float* __restrict__ b1, const float* __restrict__ b2,
    const float* __restrict__ bq, const float* __restrict__ bk,
    const float* __restrict__ bv,
    const float* __restrict__ gamma, const float* __restrict__ beta,
    const float* __restrict__ bo,
    const unsigned short* __restrict__ wf,
    float* __restrict__ out) {
  // 2 waves/block, 2 ping-pong 16KB activation buffers per wave = 64KB LDS.
  __shared__ __align__(16) unsigned short act[2][2][8192];
  const int lane = threadIdx.x & 31;
  const int wave = threadIdx.x >> 5;
  unsigned short* bufA = act[wave][0];
  unsigned short* bufB = act[wave][1];
  const int rowbase = (blockIdx.x * 2 + wave) * 16;

  // ---- stage x -> bufA: fp32 global load + bf16 pack, row-major 16x512 -----
  {
    const int r = lane & 15, hhalf = lane >> 4;
    const float* xr = x + (size_t)(rowbase + r) * 512 + hhalf * 256;
    unsigned short* dst = bufA + r * 512 + hhalf * 256;
    for (int i = 0; i < 16; ++i) {
      const f32x4* p = (const f32x4*)(xr + i * 16);
      f32x4 p0 = p[0], p1 = p[1], p2 = p[2], p3 = p[3];
      u32x4 w0, w1;
      w0.x = pk2(p0.x, p0.y); w0.y = pk2(p0.z, p0.w);
      w0.z = pk2(p1.x, p1.y); w0.w = pk2(p1.z, p1.w);
      w1.x = pk2(p2.x, p2.y); w1.y = pk2(p2.z, p2.w);
      w1.z = pk2(p3.x, p3.y); w1.w = pk2(p3.z, p3.w);
      *(u32x4*)(dst + i * 16) = w0;
      *(u32x4*)(dst + i * 16 + 8) = w1;
    }
  }

  // ---- MLP -----------------------------------------------------------------
  dense_relu<32>(wf + OFF_W1, b1, bufA, bufB, lane);   // h1 -> bufB
  dense_relu<32>(wf + OFF_W2, b2, bufB, bufA, lane);   // h2 -> bufA

  // ---- QKV + scalar gate (Q,K never materialized) --------------------------
  v8f Vt[16];
  v8f sacc = vzero();
  #pragma unroll
  for (int t = 0; t < 16; ++t) {
    v8f qa = vzero(), ka = vzero(), va = vzero();
    #pragma unroll
    for (int k = 0; k < 16; ++k) {
      v16bf a = ldA_act(bufA, k, lane, 512);
      qa = wmma_bf16(a, ldB(wf + OFF_WQ, k * 16 + t, lane), qa);
      ka = wmma_bf16(a, ldB(wf + OFF_WK, k * 16 + t, lane), ka);
      va = wmma_bf16(a, ldB(wf + OFF_WV, k * 16 + t, lane), va);
    }
    const int c = t * 16 + (lane & 15);
    const float bqv = bq[c], bkv = bk[c], bvv = bv[c];
    #pragma unroll
    for (int v = 0; v < 8; ++v) {
      float qv = qa[v] + bqv, kv = ka[v] + bkv;
      sacc[v] += qv * kv;          // row = v (+8 upper lanes), col = lane
      va[v]   += bvv;
    }
    Vt[t] = va;
  }
  // reduce scores across the 16 columns held per half-wave (xor < 16 stays
  // inside each 16-lane half)
  #pragma unroll
  for (int off = 1; off < 16; off <<= 1)
    #pragma unroll
    for (int v = 0; v < 8; ++v) sacc[v] += __shfl_xor(sacc[v], off, 32);

  v8f gate;
  #pragma unroll
  for (int v = 0; v < 8; ++v) gate[v] = 1.f / (1.f + __expf(-sacc[v]));

  // gate * V + LayerNorm statistics over 256 cols
  v8f ms = vzero(), m2 = vzero();
  #pragma unroll
  for (int t = 0; t < 16; ++t)
    #pragma unroll
    for (int v = 0; v < 8; ++v) {
      float o = gate[v] * Vt[t][v];
      Vt[t][v] = o;
      ms[v] += o;
      m2[v] += o * o;
    }
  #pragma unroll
  for (int off = 1; off < 16; off <<= 1)
    #pragma unroll
    for (int v = 0; v < 8; ++v) {
      ms[v] += __shfl_xor(ms[v], off, 32);
      m2[v] += __shfl_xor(m2[v], off, 32);
    }

  v8f mu, rs;
  #pragma unroll
  for (int v = 0; v < 8; ++v) {
    float m   = ms[v] * (1.f / 256.f);
    float var = m2[v] * (1.f / 256.f) - m * m;
    mu[v] = m;
    rs[v] = rsqrtf(var + 1e-5f);
  }

  // normalize and lay out as 16x256 A-source in bufB
  #pragma unroll
  for (int t = 0; t < 16; ++t) {
    const int c = t * 16 + (lane & 15);
    const float ga = gamma[c], be = beta[c];
    v8f o;
    #pragma unroll
    for (int v = 0; v < 8; ++v) o[v] = (Vt[t][v] - mu[v]) * rs[v] * ga + be;
    stTile_act(bufB, o, t, lane, 256);
  }

  // ---- final projection 256 -> 512, f32 store ------------------------------
  v16bf aof[8];
  #pragma unroll
  for (int k = 0; k < 8; ++k) aof[k] = ldA_act(bufB, k, lane, 256);

  const int hi8 = (lane >= 16) ? 8 : 0, n0 = lane & 15;
  for (int ct = 0; ct < 32; ++ct) {
    v16bf b[8];
    #pragma unroll
    for (int k = 0; k < 8; ++k) b[k] = ldB(wf + OFF_WO, k * 32 + ct, lane);
    v8f acc = vzero();
    #pragma unroll
    for (int k = 0; k < 8; ++k) acc = wmma_bf16(aof[k], b[k], acc);
    const float bs = bo[ct * 16 + n0];
    #pragma unroll
    for (int v = 0; v < 8; ++v)
      out[(size_t)(rowbase + v + hi8) * 512 + ct * 16 + n0] = acc[v] + bs;
  }
}

// ---------------------------------------------------------------------------
extern "C" void kernel_launch(void* const* d_in, const int* in_sizes, int n_in,
                              void* d_out, int out_size, void* d_ws, size_t ws_size,
                              hipStream_t stream) {
  const float* x     = (const float*)d_in[0];
  const float* W1    = (const float*)d_in[1];
  const float* b1    = (const float*)d_in[2];
  const float* W2    = (const float*)d_in[3];
  const float* b2    = (const float*)d_in[4];
  const float* Wq    = (const float*)d_in[5];
  const float* bq    = (const float*)d_in[6];
  const float* Wk    = (const float*)d_in[7];
  const float* bk    = (const float*)d_in[8];
  const float* Wv    = (const float*)d_in[9];
  const float* bv    = (const float*)d_in[10];
  const float* gamma = (const float*)d_in[11];
  const float* beta  = (const float*)d_in[12];
  const float* Wo    = (const float*)d_in[13];
  const float* bo    = (const float*)d_in[14];
  float* out = (float*)d_out;
  unsigned short* wf = (unsigned short*)d_ws;  // ~2MB repacked bf16 weights

  prep_w<<<dim3((512 * 512) / 256), dim3(256), 0, stream>>>(W1, wf + OFF_W1, 512, 512 * 512);
  prep_w<<<dim3((512 * 512) / 256), dim3(256), 0, stream>>>(W2, wf + OFF_W2, 512, 512 * 512);
  prep_w<<<dim3((512 * 256) / 256), dim3(256), 0, stream>>>(Wq, wf + OFF_WQ, 256, 512 * 256);
  prep_w<<<dim3((512 * 256) / 256), dim3(256), 0, stream>>>(Wk, wf + OFF_WK, 256, 512 * 256);
  prep_w<<<dim3((512 * 256) / 256), dim3(256), 0, stream>>>(Wv, wf + OFF_WV, 256, 512 * 256);
  prep_w<<<dim3((256 * 512) / 256), dim3(256), 0, stream>>>(Wo, wf + OFF_WO, 512, 256 * 512);

  // 65536 tokens / (2 waves * 16 rows) = 2048 workgroups of 64 threads
  fused_encoder<<<dim3(2048), dim3(64), 0, stream>>>(
      x, b1, b2, bq, bk, bv, gamma, beta, bo, wf, out);
}
